// ScouterAttention_18176301597397
// MI455X (gfx1250) — matrix operand order repaired and checked by reading, not compiled, not run on hardware
//
#include <hip/hip_runtime.h>
#include <hip/hip_bf16.h>

// ---------------------------------------------------------------------------
// ScouterAttention for MI455X (gfx1250): bf16 WMMA everywhere.
//   B=128, N=1024, D=512, S=64. The reference's 3-iteration loop is
//   idempotent (q never updates) -> computed once.
// ---------------------------------------------------------------------------

typedef __bf16 bf16;
typedef __attribute__((ext_vector_type(8)))  bf16  v8bf;
typedef __attribute__((ext_vector_type(16))) bf16  v16bf;
typedef __attribute__((ext_vector_type(8)))  float v8f;
typedef __attribute__((ext_vector_type(4)))  float v4f;

static constexpr int   kB  = 128;
static constexpr int   kN  = 1024;
static constexpr int   kD  = 512;
static constexpr int   kS  = 64;
static constexpr float kScale = 0.04419417382415922f;   // 512^-0.5
static constexpr float kEps   = 1e-8f;

// --- WMMA helpers -----------------------------------------------------------

__device__ __forceinline__ v8f wmma_bf16(v16bf a, v16bf b, v8f c) {
  // D = A(16x32 bf16) x B(32x16 bf16) + C(16x16 f32)
  return __builtin_amdgcn_wmma_f32_16x16x32_bf16(
      /*neg_a=*/false, a, /*neg_b=*/false, b,
      /*c_mod=*/(short)0, c, /*reuse_a=*/false, /*reuse_b=*/false);
}

__device__ __forceinline__ v16bf make16(v8bf lo, v8bf hi) {
  v16bf r;
#pragma unroll
  for (int e = 0; e < 8; ++e) { r[e] = lo[e]; r[e + 8] = hi[e]; }
  return r;
}

// B fragment: p is offset by 16*(lane>>4); lane holds K = {8*hl..8*hl+7} U
// {16+8*hl..16+8*hl+7} (matches ldA16). 2x16B loads.
__device__ __forceinline__ v16bf ldB16(const bf16* p) {
  const bf16* q = p - 8 * ((threadIdx.x & 31) >> 4);
  return make16(*(const v8bf*)q, *(const v8bf*)(q + 16));
}

// A fragment: lane holds its row's K = {8*hl..8*hl+7} U {16+8*hl..16+8*hl+7}.
// rowp points at row*pitch + k0.
__device__ __forceinline__ v16bf ldA16(const bf16* rowp, int hl) {
  return make16(*(const v8bf*)(rowp + 8 * hl),
                *(const v8bf*)(rowp + 8 * hl + 16));
}

__device__ __forceinline__ v8f zero8() {
  v8f z = {0.f, 0.f, 0.f, 0.f, 0.f, 0.f, 0.f, 0.f};
  return z;
}

// --- Kernel 0: fp32 weights -> bf16 scratch --------------------------------

__global__ void __launch_bounds__(256) cvt_weights(
    const float* __restrict__ W1, const float* __restrict__ W2,
    const float* __restrict__ W3, bf16* __restrict__ out) {
  int idx = blockIdx.x * 256 + threadIdx.x;          // 0 .. 3*512*512-1
  const float* src = (idx < 262144) ? W1 : ((idx < 524288) ? W2 : W3);
  out[idx] = (bf16)src[idx & 262143];
}

// --- Kernel 1: fused 3-layer MLP, k = L3(relu(L2(relu(L1(x))))) -------------
// One block = 64 rows. Both intermediates stay in LDS (2 x 64x520 bf16).
// 8 waves; each wave owns a 64-row x 64-col tile (4 m-tiles x 4 n-tiles) so
// every A/B fragment is reused 4x from registers: 8 loads per 16 WMMAs.

static constexpr int kMlpPitch = 520;   // bf16 elems/row; 1040B -> conflict-free

__global__ void __launch_bounds__(256) mlp_kernel(
    const float* __restrict__ X, const bf16* __restrict__ Wb,
    const float* __restrict__ b1, const float* __restrict__ b2,
    const float* __restrict__ b3, bf16* __restrict__ Kout) {
  extern __shared__ bf16 sm_mlp[];
  bf16* buf0 = sm_mlp;
  bf16* buf1 = sm_mlp + 64 * kMlpPitch;

  const int t = threadIdx.x;
  const int w = t >> 5, lane = t & 31, hl = lane >> 4, lm = lane & 15;
  const long row0 = (long)blockIdx.x * 64;

  // Stage X tile fp32 -> bf16 LDS (64x512, vec4 loads).
  const v4f* X4 = (const v4f*)(X + row0 * kD);
#pragma unroll 4
  for (int it = 0; it < 32; ++it) {
    int idx = t + it * 256;            // 8192 vec4 chunks
    int r = idx >> 7, c4 = idx & 127;
    v4f v = X4[idx];
    bf16* dst = buf0 + r * kMlpPitch + c4 * 4;
    dst[0] = (bf16)v.x; dst[1] = (bf16)v.y;
    dst[2] = (bf16)v.z; dst[3] = (bf16)v.w;
  }
  __syncthreads();

  bf16* cur = buf0;
  bf16* nxt = buf1;
  const float* biases[3] = {b1, b2, b3};

#pragma unroll 1
  for (int layer = 0; layer < 3; ++layer) {
    const bf16* W = Wb + (long)layer * kD * kD;   // row-major [out=512][in=512]
    const float* bias = biases[layer];

    v8f acc[16];                                  // acc[mi*4+nt]
#pragma unroll
    for (int q = 0; q < 16; ++q) acc[q] = zero8();

#pragma unroll 1
    for (int ks = 0; ks < 16; ++ks) {            // K = 512, step 32
      int k0 = ks * 32;
      v16bf a[4];
#pragma unroll
      for (int mi = 0; mi < 4; ++mi)
        a[mi] = ldA16(cur + (mi * 16 + lm) * kMlpPitch + k0, hl);
#pragma unroll
      for (int nt = 0; nt < 4; ++nt) {
        int o = w * 64 + nt * 16 + lm;           // output-feature = B column
        v16bf b = ldB16(W + (long)o * kD + k0 + 16 * hl);
#pragma unroll
        for (int mi = 0; mi < 4; ++mi)
          acc[mi * 4 + nt] = wmma_bf16(a[mi], b, acc[mi * 4 + nt]);
      }
    }

    // Epilogue: +bias, ReLU (layers 0/1), write LDS or global k (layer 2).
#pragma unroll
    for (int nt = 0; nt < 4; ++nt) {
      int col = w * 64 + nt * 16 + lm;
      float bv = bias[col];
#pragma unroll
      for (int mi = 0; mi < 4; ++mi)
#pragma unroll
        for (int r = 0; r < 8; ++r) {
          int m = mi * 16 + r + 8 * hl;
          float v = acc[mi * 4 + nt][r] + bv;
          if (layer < 2) {
            v = fmaxf(v, 0.f);
            nxt[m * kMlpPitch + col] = (bf16)v;
          } else {
            Kout[(row0 + m) * kD + col] = (bf16)v;
          }
        }
    }
    __syncthreads();
    bf16* tmp = cur; cur = nxt; nxt = tmp;
  }
}

// --- Kernel 2: attention scores per batch -----------------------------------
// One block per batch. Full 64x1024 dots matrix lives in LDS (256KB of the
// 320KB WGP LDS) so s_j / s_all / sigmoid / row-norm are all local.
// q A-fragments stay register-resident (128 VGPRs) for the whole kernel.

__global__ void __launch_bounds__(256) attn_kernel(
    const float* __restrict__ slots, const bf16* __restrict__ Kb,
    float* __restrict__ attn_out, bf16* __restrict__ attn2ws) {
  extern __shared__ float sm_at[];
  float* dots  = sm_at;                    // [64][1024]
  float* part  = sm_at + 64 * 1024;        // [256]
  float* s_j   = part + 256;               // [64]
  float* rsum  = s_j + 64;                 // [64]
  float* s_all = rsum + 64;                // [1]

  const int t = threadIdx.x, b = blockIdx.x;
  const int w = t >> 5, lane = t & 31, hl = lane >> 4, lm = lane & 15;
  const int mt = w & 3, ch = w >> 2;       // 4 row-tiles x 2 col-halves

  // q A-fragments (16 K-steps) held in registers; fp32 slots -> bf16.
  v16bf aq[16];
  {
    const float* qrow = slots + (mt * 16 + lm) * kD;
#pragma unroll
    for (int ks = 0; ks < 16; ++ks) {
      const float* p0 = qrow + ks * 32 + 8 * hl;
      const float* p1 = p0 + 16;
      v16bf a;
#pragma unroll
      for (int e = 0; e < 8; ++e) { a[e] = (bf16)p0[e]; a[e + 8] = (bf16)p1[e]; }
      aq[ks] = a;
    }
  }

  // dots = q . k^T * scale  (M=64, N=1024, K=512)
  const bf16* kbase = Kb + (long)b * kN * kD;
#pragma unroll 1
  for (int nt = 0; nt < 32; ++nt) {
    int n0 = ch * 512 + nt * 16;
    int j = n0 + lm;                        // B column = key index
    v8f acc = zero8();
#pragma unroll
    for (int ks = 0; ks < 16; ++ks) {
      v16bf bb = ldB16(kbase + (long)j * kD + ks * 32 + 16 * hl);
      acc = wmma_bf16(aq[ks], bb, acc);
    }
#pragma unroll
    for (int r = 0; r < 8; ++r) {
      int i = mt * 16 + r + 8 * hl;
      dots[i * kN + n0 + lm] = acc[r] * kScale;
    }
  }
  __syncthreads();

  // s_j (per-row sum) and s_all (batch total).
  const int i = t >> 2;
  float* row = dots + i * kN + (t & 3) * 256;
  float p = 0.f;
  for (int jj = 0; jj < 256; ++jj) p += row[jj];
  part[t] = p;
  __syncthreads();
  if (t < 64) s_j[t] = part[4 * t] + part[4 * t + 1] + part[4 * t + 2] + part[4 * t + 3];
  __syncthreads();
  if (t == 0) {
    float s = 0.f;
    for (int r2 = 0; r2 < 64; ++r2) s += s_j[r2];
    s_all[0] = s;
  }
  __syncthreads();

  // attn = sigmoid(dots * s_all / s_j); write output + keep in LDS.
  float fac = s_all[0] / s_j[i];
  float* aout = attn_out + (long)b * kS * kN + i * kN + (t & 3) * 256;
  float p2 = 0.f;
  for (int jj = 0; jj < 256; ++jj) {
    float x = row[jj] * fac;
    float a = 1.f / (1.f + __expf(-x));
    row[jj] = a;
    aout[jj] = a;
    p2 += a;
  }
  part[t] = p2;
  __syncthreads();
  if (t < 64) rsum[t] = part[4 * t] + part[4 * t + 1] + part[4 * t + 2] + part[4 * t + 3];
  __syncthreads();

  // attn2 (bf16, row-major) -> scratch for the updates GEMM.
  float inv = 1.f / (rsum[i] + kEps);
  bf16* a2 = attn2ws + (long)b * kS * kN + i * kN + (t & 3) * 256;
  for (int jj = 0; jj < 256; ++jj) a2[jj] = (bf16)(row[jj] * inv);
}

// --- Kernel 3: updates^T = inputs^T @ attn2^T  (per batch) ------------------
// A = transposed-inputs tile in LDS (fp32->bf16, padded pitch 40),
// B = attn2 rows in LDS (padded pitch 1032). M=512(d), N=64(i), K=1024(j).
// 4x4 tile per wave -> 4 A + 4 B fragment loads per 16 WMMAs.

static constexpr int kA2Pitch = 1032;
static constexpr int kTPitch  = 40;

__global__ void __launch_bounds__(256) updates_kernel(
    const float* __restrict__ inp, const bf16* __restrict__ attn2ws,
    float* __restrict__ upd_out) {
  extern __shared__ char smu[];
  bf16* a2 = (bf16*)smu;                                  // [64][1032]
  bf16* tT = (bf16*)(smu + (size_t)64 * kA2Pitch * 2);    // [512][40]

  const int t = threadIdx.x, b = blockIdx.x;
  const int w = t >> 5, lane = t & 31, hl = lane >> 4, lm = lane & 15;

  // Stage attn2[b] into padded LDS (8192 8-elem chunks).
  const bf16* src = attn2ws + (long)b * kS * kN;
#pragma unroll 4
  for (int it = 0; it < 32; ++it) {
    int idx = t + it * 256;
    int r = idx >> 7, c8 = idx & 127;
    *(v8bf*)(a2 + r * kA2Pitch + c8 * 8) = *(const v8bf*)(src + r * kN + c8 * 8);
  }

  v8f acc[16];
#pragma unroll
  for (int q = 0; q < 16; ++q) acc[q] = zero8();

  const float* ib = inp + (long)b * kN * kD;
#pragma unroll 1
  for (int j0 = 0; j0 < kN; j0 += 32) {
    __syncthreads();   // also covers the a2 staging on the first iteration
    // Stage inputs[b, j0:j0+32, :] transposed -> tT[d][jj] bf16.
#pragma unroll 4
    for (int it = 0; it < 64; ++it) {
      int e = t + it * 256;               // 0..16383
      int jj = e >> 9, d = e & 511;
      tT[d * kTPitch + jj] = (bf16)ib[(long)(j0 + jj) * kD + d];
    }
    __syncthreads();

#pragma unroll
    for (int mi = 0; mi < 4; ++mi) {
      int drow = w * 64 + mi * 16 + lm;   // M index = feature d
      v16bf a = ldA16(tT + drow * kTPitch, hl);
#pragma unroll
      for (int ni = 0; ni < 4; ++ni) {
        int i = ni * 16 + lm;             // N index = slot i
        v16bf bb = ldB16(a2 + i * kA2Pitch + j0 + 16 * hl);
        acc[mi * 4 + ni] = wmma_bf16(a, bb, acc[mi * 4 + ni]);
      }
    }
  }

  // Write updates[b, i, d] (D tile is [M=d][N=i]).
  float* ob = upd_out + (long)b * kS * kD;
#pragma unroll
  for (int mi = 0; mi < 4; ++mi)
#pragma unroll
    for (int ni = 0; ni < 4; ++ni)
#pragma unroll
      for (int r = 0; r < 8; ++r) {
        int d = w * 64 + mi * 16 + r + 8 * hl;
        int i = ni * 16 + lm;
        ob[i * kD + d] = acc[mi * 4 + ni][r];
      }
}

// --- Host launcher ----------------------------------------------------------

extern "C" void kernel_launch(void* const* d_in, const int* in_sizes, int n_in,
                              void* d_out, int out_size, void* d_ws, size_t ws_size,
                              hipStream_t stream) {
  const float* inputs_pe = (const float*)d_in[0];
  const float* inputs    = (const float*)d_in[1];
  const float* slots     = (const float*)d_in[2];
  const float* W1 = (const float*)d_in[3];
  const float* b1 = (const float*)d_in[4];
  const float* W2 = (const float*)d_in[5];
  const float* b2 = (const float*)d_in[6];
  const float* W3 = (const float*)d_in[7];
  const float* b3 = (const float*)d_in[8];

  // Workspace layout (bytes):
  //   [0, 1.5MB)        : W1/W2/W3 bf16
  //   [1.5MB, +128MB)   : k bf16   [B*N, D]
  //   [.., +16MB)       : attn2 bf16 [B, S, N]
  char* ws = (char*)d_ws;
  bf16* Wb  = (bf16*)ws;
  bf16* Kbf = (bf16*)(ws + (size_t)3 * kD * kD * 2);
  bf16* A2  = (bf16*)(ws + (size_t)3 * kD * kD * 2 + (size_t)kB * kN * kD * 2);

  float* upd_out  = (float*)d_out;                          // [B, S, D]
  float* attn_out = (float*)d_out + (size_t)kB * kS * kD;   // [B, S, N]

  cvt_weights<<<3 * kD * kD / 256, 256, 0, stream>>>(W1, W2, W3, Wb);

  size_t mlp_lds = (size_t)2 * 64 * kMlpPitch * sizeof(bf16);      // 130 KB
  mlp_kernel<<<kB * kN / 64, 256, mlp_lds, stream>>>(
      inputs_pe, Wb, b1, b2, b3, Kbf);

  size_t attn_lds = (size_t)(64 * 1024 + 256 + 64 + 64 + 8) * sizeof(float); // 258 KB
  attn_kernel<<<kB, 256, attn_lds, stream>>>(slots, Kbf, attn_out, A2);

  size_t upd_lds = (size_t)(64 * kA2Pitch + 512 * kTPitch) * sizeof(bf16);   // 169 KB
  updates_kernel<<<kB, 256, upd_lds, stream>>>(inputs, A2, upd_out);
}
